// ModernBertAttention_21732534517786
// MI455X (gfx1250) — matrix operand order repaired and checked by reading, not compiled, not run on hardware
//
#include <hip/hip_runtime.h>
#include <hip/hip_bf16.h>

#define BD     2
#define SD     2048
#define HIDD   768
#define NH     12
#define DHD    64
#define WIN    128
#define MTOT   (BD * SD)          // 4096
#define NTILES (HIDD / 16)        // 48
#define NEG_INF_F (-1.0e9f)

typedef __attribute__((ext_vector_type(16))) _Float16 v16h;
typedef __attribute__((ext_vector_type(8)))  float    v8f;

// ---------------------------------------------------------------------------
// WMMA helpers (CDNA5 wave32, D = A(16x32 f16) * B(32x16 f16) + C(16x16 f32))
// ---------------------------------------------------------------------------
__device__ __forceinline__ v8f wmma16(v16h a, v16h b, v8f c) {
  return __builtin_amdgcn_wmma_f32_16x16x32_f16(
      /*neg_a=*/false, a, /*neg_b=*/false, b,
      /*c_mod=*/(short)0, c, /*reuse_a=*/false, /*reuse_b=*/false);
}

// A-fragment (16x32 f16): lane l holds row (row0 + (l&15)); halves 0..7 ->
// K = k0 + 8*(l>>4) + 0..7 ; halves 8..15 -> K = k0 + 16 + 8*(l>>4) + 0..7.
// Two contiguous 16B loads per lane.
__device__ __forceinline__ v16h load_a_frag(const _Float16* base, int stride,
                                            int row0, int k0) {
  const int l = threadIdx.x & 31;
  const int g = l >> 4;
  const _Float16* p = base + (size_t)(row0 + (l & 15)) * stride + k0 + 8 * g;
  v16h a;
  ((float4*)&a)[0] = *(const float4*)(p);
  ((float4*)&a)[1] = *(const float4*)(p + 16);
  return a;
}

// B-fragment (32x16 f16): lane l holds column (n0 + (l&15)); halves 0..7 ->
// K = k0 + 8*(l>>4) + 0..7 ; halves 8..15 -> K = k0 + 16 + 8*(l>>4) + 0..7.
__device__ __forceinline__ v16h load_b_frag(const _Float16* base, int stride,
                                            int k0, int n0) {
  const int l = threadIdx.x & 31;
  const int g = l >> 4;
  const _Float16* p = base + (size_t)(k0 + 8 * g) * stride + n0 + (l & 15);
  v16h b;
  #pragma unroll
  for (int s = 0; s < 8; ++s) {
    b[s] = p[(size_t)s * stride];
    b[8 + s] = p[(size_t)(16 + s) * stride];
  }
  return b;
}

// ---------------------------------------------------------------------------
// Kernel 1: RoPE(hidden) -> qin (f16)  and  hidden -> vin (f16)
// ---------------------------------------------------------------------------
__global__ void rope_convert_kernel(const float* __restrict__ hs,
                                    _Float16* __restrict__ qin,
                                    _Float16* __restrict__ vin) {
  const int i = blockIdx.x * blockDim.x + threadIdx.x;
  if (i >= MTOT * HIDD) return;
  const int d = i & 63;                    // dim within head
  const int s = (i / HIDD) % SD;           // sequence position
  const float x = hs[i];
  vin[i] = (_Float16)x;
  const int j = d & 31;
  // inv_freq[j] = 10000^(-2j/64) = exp(-j * ln(10000)/32)
  const float freq = __expf(-(float)j * 0.2878231366f);
  const float ang = (float)s * freq;
  float sn, cs;
  __sincosf(ang, &sn, &cs);
  const float other = (d < 32) ? -hs[i + 32] : hs[i - 32];
  qin[i] = (_Float16)(x * cs + other * sn);
}

// ---------------------------------------------------------------------------
// Kernel 2: fp32 -> f16 weight conversion
// ---------------------------------------------------------------------------
__global__ void cvt_kernel(const float* __restrict__ w, _Float16* __restrict__ o,
                           int n) {
  const int i = blockIdx.x * blockDim.x + threadIdx.x;
  if (i < n) o[i] = (_Float16)w[i];
}

// ---------------------------------------------------------------------------
// Kernel 3: fused QKV projection. grid.y selects Q(0)/K(1)/V(2).
//   Q: (x*Wq + bq) * 1/8      -> [4096, 768] row-major (f16)
//   K: (x*Wk + bk)            -> transposed [B, H, DH, S]  (f16)
//   V: (x*Wv + bv)            -> [4096, 768] row-major (f16)
// One wave computes a 16x64 strip. Ping-pong pipelined with SEPARATE
// accumulator sets per phase (c* for x-phase, d* for y-phase; summed at the
// end): all 8 WMMAs per loop body are mutually independent -> no WAW chains,
// no allocator copies, no hazard NOPs.
// ---------------------------------------------------------------------------
__global__ __launch_bounds__(256) void qkv_gemm_kernel(
    const _Float16* __restrict__ qin, const _Float16* __restrict__ vin,
    const _Float16* __restrict__ wq, const _Float16* __restrict__ wk,
    const _Float16* __restrict__ wv, const float* __restrict__ bq,
    const float* __restrict__ bk, const float* __restrict__ bv,
    _Float16* __restrict__ qout, _Float16* __restrict__ ktout,
    _Float16* __restrict__ vout) {
  const int mode = blockIdx.y;
  const _Float16* X = (mode == 2) ? vin : qin;
  const _Float16* W = (mode == 0) ? wq : (mode == 1) ? wk : wv;
  const float* bias = (mode == 0) ? bq : (mode == 1) ? bk : bv;
  const float scale = (mode == 0) ? 0.125f : 1.0f;   // 1/sqrt(64)

  const int wave = threadIdx.x >> 5;
  const int tile = blockIdx.x * 8 + wave;            // 256 * 12 = 3072 strips
  const int mt = tile / (NTILES / 4);                // M tile (16 rows)
  const int ns = tile % (NTILES / 4);                // N strip (64 cols)
  const int n0 = ns * 64;
  const int m0 = mt * 16;

  v8f c0 = {0.f,0.f,0.f,0.f,0.f,0.f,0.f,0.f};
  v8f c1 = c0, c2 = c0, c3 = c0;                     // phase-x accumulators
  v8f d0 = c0, d1 = c0, d2 = c0, d3 = c0;            // phase-y accumulators

  // set 0 fragments: k = 0
  v16h xa  = load_a_frag(X, HIDD, m0, 0);
  v16h xb0 = load_b_frag(W, HIDD, 0, n0);
  v16h xb1 = load_b_frag(W, HIDD, 0, n0 + 16);
  v16h xb2 = load_b_frag(W, HIDD, 0, n0 + 32);
  v16h xb3 = load_b_frag(W, HIDD, 0, n0 + 48);

  // steady state: 64 K-elements (two WMMA steps) per iteration
  for (int k0 = 0; k0 < HIDD - 64; k0 += 64) {
    const int k1 = k0 + 32, k2 = k0 + 64;
    // prefetch set 1 (k1)
    v16h ya  = load_a_frag(X, HIDD, m0, k1);
    v16h yb0 = load_b_frag(W, HIDD, k1, n0);
    v16h yb1 = load_b_frag(W, HIDD, k1, n0 + 16);
    v16h yb2 = load_b_frag(W, HIDD, k1, n0 + 32);
    v16h yb3 = load_b_frag(W, HIDD, k1, n0 + 48);
    // consume set 0 into c*
    c0 = wmma16(xa, xb0, c0);
    c1 = wmma16(xa, xb1, c1);
    c2 = wmma16(xa, xb2, c2);
    c3 = wmma16(xa, xb3, c3);
    // prefetch set 0 (k2)
    xa  = load_a_frag(X, HIDD, m0, k2);
    xb0 = load_b_frag(W, HIDD, k2, n0);
    xb1 = load_b_frag(W, HIDD, k2, n0 + 16);
    xb2 = load_b_frag(W, HIDD, k2, n0 + 32);
    xb3 = load_b_frag(W, HIDD, k2, n0 + 48);
    // consume set 1 into d*
    d0 = wmma16(ya, yb0, d0);
    d1 = wmma16(ya, yb1, d1);
    d2 = wmma16(ya, yb2, d2);
    d3 = wmma16(ya, yb3, d3);
  }
  // tail pair: k = HIDD-64 (in set 0) and k = HIDD-32
  {
    const int k1 = HIDD - 32;
    v16h ya  = load_a_frag(X, HIDD, m0, k1);
    v16h yb0 = load_b_frag(W, HIDD, k1, n0);
    v16h yb1 = load_b_frag(W, HIDD, k1, n0 + 16);
    v16h yb2 = load_b_frag(W, HIDD, k1, n0 + 32);
    v16h yb3 = load_b_frag(W, HIDD, k1, n0 + 48);
    c0 = wmma16(xa, xb0, c0);
    c1 = wmma16(xa, xb1, c1);
    c2 = wmma16(xa, xb2, c2);
    c3 = wmma16(xa, xb3, c3);
    d0 = wmma16(ya, yb0, d0);
    d1 = wmma16(ya, yb1, d1);
    d2 = wmma16(ya, yb2, d2);
    d3 = wmma16(ya, yb3, d3);
  }
  // merge phases
  c0 += d0; c1 += d1; c2 += d2; c3 += d3;

  const int l = threadIdx.x & 31;
  const int g = l >> 4, nl = l & 15;
  #pragma unroll
  for (int f = 0; f < 4; ++f) {
    const v8f c = (f == 0) ? c0 : (f == 1) ? c1 : (f == 2) ? c2 : c3;
    const int n = n0 + f * 16 + nl;
    const float bval = bias[n];
    #pragma unroll
    for (int r = 0; r < 8; ++r) {
      const int m = m0 + r + 8 * g;                  // global row (b*S + s)
      const _Float16 hv = (_Float16)((c[r] + bval) * scale);
      if (mode == 1) {
        const int bb = m >> 11, s = m & (SD - 1);
        const int hh = n >> 6, d = n & 63;
        ktout[((size_t)((bb * NH + hh) * DHD + d) << 11) + s] = hv;
      } else if (mode == 0) {
        qout[(size_t)m * HIDD + n] = hv;
      } else {
        vout[(size_t)m * HIDD + n] = hv;
      }
    }
  }
}

// ---------------------------------------------------------------------------
// Kernel 4: sliding-window flash attention. One wave = one (b, h, 16-query)
// tile. 32-key chunks: 4 WMMAs for S = Q*K^T, online softmax (row stats via
// half-wave shfl_xor reductions), P transposed via per-wave LDS to A-layout,
// 4 WMMAs for O += P*V.
// ---------------------------------------------------------------------------
__global__ __launch_bounds__(128) void attn_kernel(
    const _Float16* __restrict__ qh, const _Float16* __restrict__ kt,
    const _Float16* __restrict__ vh, const int* __restrict__ amask,
    _Float16* __restrict__ ao) {
  __shared__ __align__(16) _Float16 lds_p[4][16 * 32];

  const int wave = threadIdx.x >> 5;
  const int gw = blockIdx.x * 4 + wave;              // 3072 waves total
  const int qt = gw & 127;
  const int bh = gw >> 7;
  const int b = bh / NH, h = bh % NH;
  const int q0 = qt * 16;
  const int l = threadIdx.x & 31;
  const int g = l >> 4, nl = l & 15;

  const _Float16* qbase = qh + h * DHD;                            // [4096,768]
  const _Float16* ktb = kt + (size_t)((b * NH + h) * DHD) * SD;    // [DH, S]
  const _Float16* vb = vh + (size_t)b * SD * HIDD + h * DHD;       // [S, 768]

  const v16h qa0 = load_a_frag(qbase, HIDD, b * SD + q0, 0);
  const v16h qa1 = load_a_frag(qbase, HIDD, b * SD + q0, 32);

  float mrow[8], lrow[8];
  v8f o0 = {0.f,0.f,0.f,0.f,0.f,0.f,0.f,0.f};
  v8f o1 = o0, o2 = o0, o3 = o0;
  #pragma unroll
  for (int r = 0; r < 8; ++r) { mrow[r] = -1.0e30f; lrow[r] = 0.f; }

  int kstart = q0 - (WIN - 1); if (kstart < 0) kstart = 0;
  kstart &= ~31;                                      // 32-aligned chunks
  int kend = q0 + 16 + (WIN - 1); if (kend > SD) kend = SD;

  _Float16* lp = &lds_p[wave][0];

  for (int kk = kstart; kk < kend; kk += 32) {
    // ---- scores: 16 queries x 32 keys ----
    v8f s0 = {0.f,0.f,0.f,0.f,0.f,0.f,0.f,0.f};
    v8f s1 = s0;
    {
      v16h b0 = load_b_frag(ktb, SD, 0, kk);
      v16h b1 = load_b_frag(ktb, SD, 32, kk);
      s0 = wmma16(qa0, b0, s0);
      s0 = wmma16(qa1, b1, s0);
    }
    {
      v16h b0 = load_b_frag(ktb, SD, 0, kk + 16);
      v16h b1 = load_b_frag(ktb, SD, 32, kk + 16);
      s1 = wmma16(qa0, b0, s1);
      s1 = wmma16(qa1, b1, s1);
    }
    // ---- mask: pad & |q-k| < WINDOW ----
    const int key0 = kk + nl, key1 = kk + 16 + nl;
    const bool pad0 = amask[b * SD + key0] != 0;
    const bool pad1 = amask[b * SD + key1] != 0;
    #pragma unroll
    for (int r = 0; r < 8; ++r) {
      const int row = q0 + r + 8 * g;
      int d0 = row - key0; if (d0 < 0) d0 = -d0;
      int d1 = row - key1; if (d1 < 0) d1 = -d1;
      if (!(pad0 && d0 < WIN)) s0[r] = NEG_INF_F;
      if (!(pad1 && d1 < WIN)) s1[r] = NEG_INF_F;
    }
    // ---- online softmax (rows live across 16-lane half groups) ----
    #pragma unroll
    for (int r = 0; r < 8; ++r) {
      float cm = fmaxf(s0[r], s1[r]);
      for (int msk = 1; msk < 16; msk <<= 1)
        cm = fmaxf(cm, __shfl_xor(cm, msk, 32));
      const float mnew = fmaxf(mrow[r], cm);
      const float alpha = __expf(mrow[r] - mnew);
      const float p0 = __expf(s0[r] - mnew);
      const float p1 = __expf(s1[r] - mnew);
      s0[r] = p0; s1[r] = p1;
      float rs = p0 + p1;
      for (int msk = 1; msk < 16; msk <<= 1)
        rs += __shfl_xor(rs, msk, 32);
      lrow[r] = lrow[r] * alpha + rs;
      mrow[r] = mnew;
      o0[r] *= alpha; o1[r] *= alpha; o2[r] *= alpha; o3[r] *= alpha;
    }
    // ---- transpose P (C-layout -> A-layout) through wave-private LDS ----
    #pragma unroll
    for (int r = 0; r < 8; ++r) {
      const int row = r + 8 * g;
      lp[row * 32 + nl]      = (_Float16)s0[r];
      lp[row * 32 + 16 + nl] = (_Float16)s1[r];
    }
    const v16h pa = load_a_frag(lp, 32, 0, 0);
    // ---- O += P * V  (16x32 @ 32x64) ----
    { v16h v0 = load_b_frag(vb, HIDD, kk, 0);  o0 = wmma16(pa, v0, o0); }
    { v16h v1 = load_b_frag(vb, HIDD, kk, 16); o1 = wmma16(pa, v1, o1); }
    { v16h v2 = load_b_frag(vb, HIDD, kk, 32); o2 = wmma16(pa, v2, o2); }
    { v16h v3 = load_b_frag(vb, HIDD, kk, 48); o3 = wmma16(pa, v3, o3); }
  }

  // ---- normalize and store [B,S,H,DH] as f16 ----
  #pragma unroll
  for (int r = 0; r < 8; ++r) {
    const int row = b * SD + q0 + r + 8 * g;
    const float inv = 1.0f / lrow[r];
    _Float16* op = ao + (size_t)row * HIDD + h * DHD + nl;
    op[0]  = (_Float16)(o0[r] * inv);
    op[16] = (_Float16)(o1[r] * inv);
    op[32] = (_Float16)(o2[r] * inv);
    op[48] = (_Float16)(o3[r] * inv);
  }
}

// ---------------------------------------------------------------------------
// Kernel 5: output projection  out = ao(f16) * Wo(f16) + bo  -> f32
// 16x64 strip per wave, same dual-accumulator ping-pong scheme as kernel 3.
// ---------------------------------------------------------------------------
__global__ __launch_bounds__(256) void out_proj_kernel(
    const _Float16* __restrict__ ao, const _Float16* __restrict__ wo,
    const float* __restrict__ bo, float* __restrict__ out) {
  const int wave = threadIdx.x >> 5;
  const int tile = blockIdx.x * 8 + wave;            // 3072 strips
  const int mt = tile / (NTILES / 4);
  const int ns = tile % (NTILES / 4);
  const int n0 = ns * 64;
  const int m0 = mt * 16;

  v8f c0 = {0.f,0.f,0.f,0.f,0.f,0.f,0.f,0.f};
  v8f c1 = c0, c2 = c0, c3 = c0;
  v8f d0 = c0, d1 = c0, d2 = c0, d3 = c0;

  v16h xa  = load_a_frag(ao, HIDD, m0, 0);
  v16h xb0 = load_b_frag(wo, HIDD, 0, n0);
  v16h xb1 = load_b_frag(wo, HIDD, 0, n0 + 16);
  v16h xb2 = load_b_frag(wo, HIDD, 0, n0 + 32);
  v16h xb3 = load_b_frag(wo, HIDD, 0, n0 + 48);

  for (int k0 = 0; k0 < HIDD - 64; k0 += 64) {
    const int k1 = k0 + 32, k2 = k0 + 64;
    v16h ya  = load_a_frag(ao, HIDD, m0, k1);
    v16h yb0 = load_b_frag(wo, HIDD, k1, n0);
    v16h yb1 = load_b_frag(wo, HIDD, k1, n0 + 16);
    v16h yb2 = load_b_frag(wo, HIDD, k1, n0 + 32);
    v16h yb3 = load_b_frag(wo, HIDD, k1, n0 + 48);
    c0 = wmma16(xa, xb0, c0);
    c1 = wmma16(xa, xb1, c1);
    c2 = wmma16(xa, xb2, c2);
    c3 = wmma16(xa, xb3, c3);
    xa  = load_a_frag(ao, HIDD, m0, k2);
    xb0 = load_b_frag(wo, HIDD, k2, n0);
    xb1 = load_b_frag(wo, HIDD, k2, n0 + 16);
    xb2 = load_b_frag(wo, HIDD, k2, n0 + 32);
    xb3 = load_b_frag(wo, HIDD, k2, n0 + 48);
    d0 = wmma16(ya, yb0, d0);
    d1 = wmma16(ya, yb1, d1);
    d2 = wmma16(ya, yb2, d2);
    d3 = wmma16(ya, yb3, d3);
  }
  {
    const int k1 = HIDD - 32;
    v16h ya  = load_a_frag(ao, HIDD, m0, k1);
    v16h yb0 = load_b_frag(wo, HIDD, k1, n0);
    v16h yb1 = load_b_frag(wo, HIDD, k1, n0 + 16);
    v16h yb2 = load_b_frag(wo, HIDD, k1, n0 + 32);
    v16h yb3 = load_b_frag(wo, HIDD, k1, n0 + 48);
    c0 = wmma16(xa, xb0, c0);
    c1 = wmma16(xa, xb1, c1);
    c2 = wmma16(xa, xb2, c2);
    c3 = wmma16(xa, xb3, c3);
    d0 = wmma16(ya, yb0, d0);
    d1 = wmma16(ya, yb1, d1);
    d2 = wmma16(ya, yb2, d2);
    d3 = wmma16(ya, yb3, d3);
  }
  c0 += d0; c1 += d1; c2 += d2; c3 += d3;

  const int l = threadIdx.x & 31;
  const int g = l >> 4, nl = l & 15;
  #pragma unroll
  for (int f = 0; f < 4; ++f) {
    const v8f c = (f == 0) ? c0 : (f == 1) ? c1 : (f == 2) ? c2 : c3;
    const int n = n0 + f * 16 + nl;
    const float bval = bo[n];
    #pragma unroll
    for (int r = 0; r < 8; ++r) {
      const int m = m0 + r + 8 * g;
      out[(size_t)m * HIDD + n] = c[r] + bval;
    }
  }
}

// ---------------------------------------------------------------------------
// Host launcher
// ---------------------------------------------------------------------------
extern "C" void kernel_launch(void* const* d_in, const int* in_sizes, int n_in,
                              void* d_out, int out_size, void* d_ws,
                              size_t ws_size, hipStream_t stream) {
  const float* hs   = (const float*)d_in[0];
  const int* amask  = (const int*)d_in[1];
  const float* Wq   = (const float*)d_in[2];
  const float* bq   = (const float*)d_in[3];
  const float* Wk   = (const float*)d_in[4];
  const float* bk   = (const float*)d_in[5];
  const float* Wv   = (const float*)d_in[6];
  const float* bv   = (const float*)d_in[7];
  const float* Wo   = (const float*)d_in[8];
  const float* bo   = (const float*)d_in[9];
  float* out = (float*)d_out;

  const size_t NE = (size_t)MTOT * HIDD;   // 3,145,728 halves per activation
  const size_t WE = (size_t)HIDD * HIDD;   //   589,824 halves per weight
  _Float16* ws  = (_Float16*)d_ws;
  _Float16* qin = ws;            // roped input (f16)
  _Float16* vin = qin + NE;      // raw input   (f16)
  _Float16* qhp = vin + NE;      // Q projected
  _Float16* ktp = qhp + NE;      // K projected, transposed [B,H,DH,S]
  _Float16* vhp = ktp + NE;      // V projected
  _Float16* aop = vhp + NE;      // attention output (pre-Wo)
  _Float16* wqh = aop + NE;
  _Float16* wkh = wqh + WE;
  _Float16* wvh = wkh + WE;
  _Float16* woh = wvh + WE;

  rope_convert_kernel<<<(unsigned)((NE + 255) / 256), 256, 0, stream>>>(hs, qin, vin);
  cvt_kernel<<<(unsigned)((WE + 255) / 256), 256, 0, stream>>>(Wq, wqh, (int)WE);
  cvt_kernel<<<(unsigned)((WE + 255) / 256), 256, 0, stream>>>(Wk, wkh, (int)WE);
  cvt_kernel<<<(unsigned)((WE + 255) / 256), 256, 0, stream>>>(Wv, wvh, (int)WE);
  cvt_kernel<<<(unsigned)((WE + 255) / 256), 256, 0, stream>>>(Wo, woh, (int)WE);

  // 4096/16 row-tiles * 768/64 col-strips = 3072 strips, 8 waves/block
  qkv_gemm_kernel<<<dim3(384, 3), 256, 0, stream>>>(
      qin, vin, wqh, wkh, wvh, bq, bk, bv, qhp, ktp, vhp);

  // 2*12*128 = 3072 q-tiles, 4 waves/block -> 768 blocks
  attn_kernel<<<768, 128, 0, stream>>>(qhp, ktp, vhp, amask, aop);

  out_proj_kernel<<<384, 256, 0, stream>>>(aop, woh, bo, out);
}
